// HGRNFloatingPoint_12386685681970
// MI455X (gfx1250) — compile-verified, matrix operand order checked by reading, not run on hardware
//
#include <hip/hip_runtime.h>
#include <hip/hip_bf16.h>
#include <cstdint>

typedef __attribute__((ext_vector_type(16))) __bf16 v16bf;
typedef __attribute__((ext_vector_type(8)))  __bf16 v8bf;
typedef __attribute__((ext_vector_type(8)))  float  v8f;

// Problem sizes (reference: B=4, S=2048, H=1024)
constexpr int Bb = 4, Ss = 2048, Hh = 1024;
constexpr int Mm = Bb * Ss;          // 8192
constexpr int Kk = Hh;               // 1024
constexpr int Nn = Hh;               // 1024

// GEMM tiling: workgroup tile 128(M) x 64(N) x 32(K), 8 waves (4M x 2N),
// wave tile 32x32 = 2x2 of 16x16x32 WMMA tiles, 3xBF16 split accumulation.
constexpr int BM = 128, BN = 64, BK = 32;
constexpr int LDT = BK + 8;          // 40 elems = 80B pitch -> conflict-free ds_load_b128

// Chunked scan
constexpr int CH = 8;                // chunks along S
constexpr int CL = Ss / CH;          // 256

// Split fp32 into hi+lo bf16 (round-to-nearest-ish). a ~= hi + lo to ~2^-18 rel.
__device__ __forceinline__ void split_bf16(float a, unsigned short& hi, unsigned short& lo) {
    unsigned int b  = __float_as_uint(a);
    unsigned int hb = (b + 0x8000u) & 0xFFFF0000u;
    float fhi = __uint_as_float(hb);
    float r   = a - fhi;                           // exact (Sterbenz)
    unsigned int rb = __float_as_uint(r) + 0x8000u;
    hi = (unsigned short)(hb >> 16);
    lo = (unsigned short)(rb >> 16);
}

// One-shot conversion: fp32 plane -> separate hi/lo bf16 planes (float4 per thread).
__global__ void split_convert_kernel(const float* __restrict__ in,
                                     unsigned short* __restrict__ ohi,
                                     unsigned short* __restrict__ olo) {
    const size_t e = (size_t)blockIdx.x * blockDim.x + threadIdx.x;  // float4 units
    const float4 v = reinterpret_cast<const float4*>(in)[e];
    unsigned short h0,h1,h2,h3,l0,l1,l2,l3;
    split_bf16(v.x, h0, l0); split_bf16(v.y, h1, l1);
    split_bf16(v.z, h2, l2); split_bf16(v.w, h3, l3);
    uint2 ph; ph.x = (unsigned)h0 | ((unsigned)h1 << 16); ph.y = (unsigned)h2 | ((unsigned)h3 << 16);
    uint2 pl; pl.x = (unsigned)l0 | ((unsigned)l1 << 16); pl.y = (unsigned)l2 | ((unsigned)l3 << 16);
    reinterpret_cast<uint2*>(ohi)[e] = ph;
    reinterpret_cast<uint2*>(olo)[e] = pl;
}

union FragU { v16bf v; v8bf h[2]; };

// MODE 0: fused projection GEMM vs 3 pre-split weights (i,f,g):
//   epilogue: O0 = sigmoid(f), O1 = i*sigmoid(i)*(1-sigmoid(f)), O2 = g
// MODE 1: plain GEMM vs 1 pre-split weight: O0 = A@W^T
// Whi/Wlo hold NW weights concatenated: [w][N][K].
template <int MODE>
__global__ __launch_bounds__(256, 1) void gemm_bf16s_kernel(
    const unsigned short* __restrict__ Ahi, const unsigned short* __restrict__ Alo,
    const unsigned short* __restrict__ Whi, const unsigned short* __restrict__ Wlo,
    float* __restrict__ O0, float* __restrict__ O1, float* __restrict__ O2)
{
    constexpr int NW = (MODE == 0) ? 3 : 1;
    struct alignas(16) Smem {
        unsigned short a_hi[BM][LDT];
        unsigned short a_lo[BM][LDT];
        unsigned short b_hi[NW][BN][LDT];
        unsigned short b_lo[NW][BN][LDT];
    };
    __shared__ Smem sm;

    const int tid   = threadIdx.x;
    const int lane  = tid & 31;
    const int wave  = tid >> 5;
    const int wm    = wave & 3;          // wave M index 0..3
    const int wn    = wave >> 2;         // wave N index 0..1
    const int lo16  = lane & 15;
    const int khalf = lane >> 4;         // 0|1
    const int kbase = khalf * 8;         // K sub-offset per CDNA5 16-bit A/B layout
    const int rsh   = khalf * 8;         // C/D row offset for upper half-wave

    const int rowM0 = blockIdx.y * BM;
    const int colN0 = blockIdx.x * BN;

    // staging assignments
    const int arow = tid >> 1;           // 0..127
    const int acol = (tid & 1) * 16;     // 0 | 16
    const int brow = tid >> 2;           // 0..63
    const int bcol = (tid & 3) * 8;      // 0,8,16,24

    const size_t aRowBase = (size_t)(rowM0 + arow) * Kk;  // + kc*BK + acol
    size_t bRowBase[NW];
#pragma unroll
    for (int w = 0; w < NW; ++w)
        bRowBase[w] = (size_t)w * Nn * Kk + (size_t)(colN0 + brow) * Kk;

    v8f acc[NW][2][2];
    const v8f vzero = {0.f, 0.f, 0.f, 0.f, 0.f, 0.f, 0.f, 0.f};
#pragma unroll
    for (int w = 0; w < NW; ++w)
#pragma unroll
        for (int mt = 0; mt < 2; ++mt)
#pragma unroll
            for (int nt = 0; nt < 2; ++nt) acc[w][mt][nt] = vzero;

    constexpr int NKC = Kk / BK;
    for (int kc = 0; kc < NKC; ++kc) {
        __syncthreads();
        // ---- stage A tile 128x32 (hi+lo bf16) ----
        {
            const size_t g = aRowBase + (size_t)kc * BK + acol;
            const uint4 h0 = *reinterpret_cast<const uint4*>(Ahi + g);
            const uint4 h1 = *reinterpret_cast<const uint4*>(Ahi + g + 8);
            const uint4 l0 = *reinterpret_cast<const uint4*>(Alo + g);
            const uint4 l1 = *reinterpret_cast<const uint4*>(Alo + g + 8);
            *reinterpret_cast<uint4*>(&sm.a_hi[arow][acol])     = h0;
            *reinterpret_cast<uint4*>(&sm.a_hi[arow][acol + 8]) = h1;
            *reinterpret_cast<uint4*>(&sm.a_lo[arow][acol])     = l0;
            *reinterpret_cast<uint4*>(&sm.a_lo[arow][acol + 8]) = l1;
            if (kc + 1 < NKC) {   // prefetch next K-slab (global_prefetch_b8)
                __builtin_prefetch(Ahi + g + BK, 0, 3);
                __builtin_prefetch(Alo + g + BK, 0, 3);
            }
        }
        // ---- stage B tiles (NW x 64 x 32) ----
#pragma unroll
        for (int w = 0; w < NW; ++w) {
            const size_t g = bRowBase[w] + (size_t)kc * BK + bcol;
            const uint4 h = *reinterpret_cast<const uint4*>(Whi + g);
            const uint4 l = *reinterpret_cast<const uint4*>(Wlo + g);
            *reinterpret_cast<uint4*>(&sm.b_hi[w][brow][bcol]) = h;
            *reinterpret_cast<uint4*>(&sm.b_lo[w][brow][bcol]) = l;
            if (kc + 1 < NKC) {
                __builtin_prefetch(Whi + g + BK, 0, 3);
                __builtin_prefetch(Wlo + g + BK, 0, 3);
            }
        }
        __syncthreads();

        // ---- A fragments (2 M-tiles) ----
        v16bf afh[2], afl[2];
#pragma unroll
        for (int mt = 0; mt < 2; ++mt) {
            const int r = wm * 32 + mt * 16 + lo16;
            FragU fh, fl;
            fh.h[0] = *reinterpret_cast<const v8bf*>(&sm.a_hi[r][kbase]);
            fh.h[1] = *reinterpret_cast<const v8bf*>(&sm.a_hi[r][16 + kbase]);
            fl.h[0] = *reinterpret_cast<const v8bf*>(&sm.a_lo[r][kbase]);
            fl.h[1] = *reinterpret_cast<const v8bf*>(&sm.a_lo[r][16 + kbase]);
            afh[mt] = fh.v; afl[mt] = fl.v;
        }
        // ---- B fragments + 3xBF16 WMMA ----
#pragma unroll
        for (int w = 0; w < NW; ++w) {
#pragma unroll
            for (int nt = 0; nt < 2; ++nt) {
                const int c = wn * 32 + nt * 16 + lo16;
                FragU bh, bl;
                bh.h[0] = *reinterpret_cast<const v8bf*>(&sm.b_hi[w][c][kbase]);
                bh.h[1] = *reinterpret_cast<const v8bf*>(&sm.b_hi[w][c][16 + kbase]);
                bl.h[0] = *reinterpret_cast<const v8bf*>(&sm.b_lo[w][c][kbase]);
                bl.h[1] = *reinterpret_cast<const v8bf*>(&sm.b_lo[w][c][16 + kbase]);
#pragma unroll
                for (int mt = 0; mt < 2; ++mt) {
                    v8f a = acc[w][mt][nt];
                    a = __builtin_amdgcn_wmma_f32_16x16x32_bf16(false, afh[mt], false, bh.v, (short)0, a, false, false);
                    a = __builtin_amdgcn_wmma_f32_16x16x32_bf16(false, afh[mt], false, bl.v, (short)0, a, false, false);
                    a = __builtin_amdgcn_wmma_f32_16x16x32_bf16(false, afl[mt], false, bh.v, (short)0, a, false, false);
                    acc[w][mt][nt] = a;
                }
            }
        }
    }

    // ---- epilogue: C/D layout: VGPR r -> (M = r + 8*khalf, N = lo16) ----
#pragma unroll
    for (int mt = 0; mt < 2; ++mt) {
#pragma unroll
        for (int nt = 0; nt < 2; ++nt) {
#pragma unroll
            for (int r = 0; r < 8; ++r) {
                const int grow = rowM0 + wm * 32 + mt * 16 + rsh + r;
                const int gcol = colN0 + wn * 32 + nt * 16 + lo16;
                const size_t idx = (size_t)grow * Nn + gcol;
                if (MODE == 0) {
                    const float iv = acc[0][mt][nt][r];
                    const float fv = acc[1][mt][nt][r];
                    const float gv = acc[2][mt][nt][r];
                    const float fs = 1.0f / (1.0f + __expf(-fv));
                    const float is = 1.0f / (1.0f + __expf(-iv));
                    O0[idx] = fs;                       // f_sig
                    O1[idx] = iv * is * (1.0f - fs);    // i_eff
                    O2[idx] = gv;                       // g
                } else {
                    O0[idx] = acc[0][mt][nt][r];
                }
            }
        }
    }
}

// Phase A: per-chunk local scan, in place: ws_i <- h' (scan with h0=0), ws_f <- cumprod(f)
__global__ void scan_chunk_kernel(float* __restrict__ ws_f, float* __restrict__ ws_i) {
    const int tid = blockIdx.x * blockDim.x + threadIdx.x;   // B*CH*H threads
    const int hh = tid % Hh;
    const int c  = (tid / Hh) % CH;
    const int b  = tid / (Hh * CH);
    size_t idx = ((size_t)b * Ss + (size_t)c * CL) * Hh + hh;
    float hp = 0.0f, cf = 1.0f;
    for (int t = 0; t < CL; ++t) {
        const float f = ws_f[idx];
        const float i = ws_i[idx];
        hp = __builtin_fmaf(f, hp, i);
        cf *= f;
        ws_i[idx] = hp;
        ws_f[idx] = cf;
        idx += Hh;
    }
}

// Phase B: serial combine across the CH chunk boundaries, seeded with h_init.
__global__ void scan_carry_kernel(const float* __restrict__ ws_f, const float* __restrict__ ws_i,
                                  const float* __restrict__ h_init, float* __restrict__ carries) {
    const int tid = blockIdx.x * blockDim.x + threadIdx.x;   // B*H threads
    const int hh = tid % Hh;
    const int b  = tid / Hh;
    float carry = h_init[(size_t)b * Hh + hh];
#pragma unroll
    for (int c = 0; c < CH; ++c) {
        carries[(size_t)c * (Bb * Hh) + (size_t)b * Hh + hh] = carry;
        const size_t idx = ((size_t)b * Ss + (size_t)c * CL + (CL - 1)) * Hh + hh;
        carry = __builtin_fmaf(ws_f[idx], carry, ws_i[idx]);   // h_end = h'_end + cf_end*carry
    }
}

// Phase C: h = h' + cf*carry ; gh = g*h, written in place over g (float4 vectorized).
__global__ void fix_gh_kernel(const float* __restrict__ ws_f, const float* __restrict__ ws_i,
                              float* __restrict__ ws_g, const float* __restrict__ carries) {
    const size_t e = (size_t)blockIdx.x * blockDim.x + threadIdx.x;  // B*S*H/4 float4's
    const int h4 = Hh / 4;
    const int hq = (int)(e % h4);
    const int t  = (int)((e / h4) % Ss);
    const int b  = (int)(e / ((size_t)h4 * Ss));
    const int c  = t / CL;
    const float4 hp = reinterpret_cast<const float4*>(ws_i)[e];
    const float4 cf = reinterpret_cast<const float4*>(ws_f)[e];
    const float4 g  = reinterpret_cast<const float4*>(ws_g)[e];
    const float4 cr = reinterpret_cast<const float4*>(carries)[(size_t)c * (Bb * h4) + (size_t)b * h4 + hq];
    float4 o;
    o.x = g.x * __builtin_fmaf(cf.x, cr.x, hp.x);
    o.y = g.y * __builtin_fmaf(cf.y, cr.y, hp.y);
    o.z = g.z * __builtin_fmaf(cf.z, cr.z, hp.z);
    o.w = g.w * __builtin_fmaf(cf.w, cr.w, hp.w);
    reinterpret_cast<float4*>(ws_g)[e] = o;
}

extern "C" void kernel_launch(void* const* d_in, const int* in_sizes, int n_in,
                              void* d_out, int out_size, void* d_ws, size_t ws_size,
                              hipStream_t stream) {
    const float* x    = (const float*)d_in[0];
    const float* h0   = (const float*)d_in[1];
    const float* w_i  = (const float*)d_in[2];
    const float* w_f  = (const float*)d_in[3];
    const float* w_g  = (const float*)d_in[4];
    const float* w_o  = (const float*)d_in[5];
    float* out = (float*)d_out;

    float* ws = (float*)d_ws;
    const size_t sz = (size_t)Mm * Hh;
    float* ws_f    = ws;                    // f_sig -> cumF
    float* ws_i    = ws + sz;               // i_eff -> h'
    float* ws_g    = ws + 2 * sz;           // g     -> g*h
    float* carries = ws + 3 * sz;           // CH * B * H floats

    unsigned short* xhi = (unsigned short*)(carries + (size_t)CH * Bb * Hh);
    unsigned short* xlo = xhi + (size_t)Mm * Kk;
    unsigned short* whi = xlo + (size_t)Mm * Kk;            // 4 weights concat [w][N][K]
    unsigned short* wlo = whi + (size_t)4 * Nn * Kk;

    const int cb = 256;
    // --- pre-split to bf16 hi/lo planes ---
    split_convert_kernel<<<(unsigned)((size_t)Mm * Kk / 4 / cb), cb, 0, stream>>>(x, xhi, xlo);
    const size_t wq = (size_t)Nn * Kk;
    split_convert_kernel<<<(unsigned)(wq / 4 / cb), cb, 0, stream>>>(w_i, whi + 0 * wq, wlo + 0 * wq);
    split_convert_kernel<<<(unsigned)(wq / 4 / cb), cb, 0, stream>>>(w_f, whi + 1 * wq, wlo + 1 * wq);
    split_convert_kernel<<<(unsigned)(wq / 4 / cb), cb, 0, stream>>>(w_g, whi + 2 * wq, wlo + 2 * wq);
    split_convert_kernel<<<(unsigned)(wq / 4 / cb), cb, 0, stream>>>(w_o, whi + 3 * wq, wlo + 3 * wq);

    dim3 gg(Nn / BN, Mm / BM);              // (16, 64)
    // --- fused projection GEMM (i,f,g) + gate epilogue ---
    gemm_bf16s_kernel<0><<<gg, cb, 0, stream>>>(xhi, xlo, whi, wlo, ws_f, ws_i, ws_g);
    // --- chunked linear scan ---
    scan_chunk_kernel<<<(Bb * CH * Hh) / cb, cb, 0, stream>>>(ws_f, ws_i);
    scan_carry_kernel<<<(Bb * Hh) / cb, cb, 0, stream>>>(ws_f, ws_i, h0, carries);
    fix_gh_kernel<<<(unsigned)((size_t)Bb * Ss * Hh / 4 / cb), cb, 0, stream>>>(ws_f, ws_i, ws_g, carries);
    // --- (g*h) needs bf16 split for the output GEMM ---
    split_convert_kernel<<<(unsigned)((size_t)Mm * Kk / 4 / cb), cb, 0, stream>>>(ws_g, xhi, xlo);
    gemm_bf16s_kernel<1><<<gg, cb, 0, stream>>>(xhi, xlo, whi + 3 * wq, wlo + 3 * wq, out, nullptr, nullptr);
}
